// DecoderRNN_16552803959307
// MI455X (gfx1250) — compile-verified, matrix operand order checked by reading, not run on hardware
//
#include <hip/hip_runtime.h>
#include <hip/hip_bf16.h>

// MI455X / gfx1250, wave32. WMMA f16 16x16x32 + async global->LDS staging.

typedef __attribute__((ext_vector_type(16))) _Float16 v16h;
typedef __attribute__((ext_vector_type(8)))  float    v8f;
typedef __attribute__((ext_vector_type(4)))  int      v4i;

#define Bsz   256
#define LATd  512
#define Hd    1024
#define Od    128
#define Tt    100
#define KC_A  64   // 2048 / 32 k-chunks for [relu(x)|h] concat
#define KC_H  32   // 1024 / 32 k-chunks

// Inverse of the ISA 16-bit fragment K-mapping (CDNA5 ISA §7.12.2):
// a 16(m)x32(k) tile is stored as 32 lanes x 16 contiguous f16 (1 KB) so that a
// lane's fragment is one 32B contiguous load.
__device__ __forceinline__ int frag_idx(int m, int k, int kchunks) {
  int mi = m >> 4, kc = k >> 5, kk = k & 31, r = m & 15;
  int hi = (kk >> 3) & 1;
  int lane = r + (hi << 4);
  int rem = kk - (hi << 3);
  int e = (rem < 8) ? rem : (rem - 8);
  return ((mi * kchunks + kc) * 32 + lane) * 16 + e;
}

__device__ __forceinline__ float sigmoidf_(float x) { return 1.0f / (1.0f + expf(-x)); }
__device__ __forceinline__ float lrelu_(float x) { return x > 0.f ? x : 0.01f * x; }

// ---- CDNA5 async global->LDS copy (ASYNCcnt path), 16B per lane ----
// Probe-confirmed signature: (v4i addrspace(1)*, v4i addrspace(3)*, imm offset, imm cpol)
__device__ __forceinline__ void async_copy16(const _Float16* gsrc, _Float16* ldst) {
#if __has_builtin(__builtin_amdgcn_global_load_async_to_lds_b128)
  __builtin_amdgcn_global_load_async_to_lds_b128(
      (__attribute__((address_space(1))) v4i*)gsrc,
      (__attribute__((address_space(3))) v4i*)ldst, 0, 0);
#else
  unsigned laddr = (unsigned)(unsigned long long)(__attribute__((address_space(3))) void*)ldst;
  asm volatile("global_load_async_to_lds_b128 %0, %1, off"
               :: "v"(laddr), "v"(gsrc) : "memory");
#endif
}

template <int N>
__device__ __forceinline__ void wait_async() {
#if __has_builtin(__builtin_amdgcn_s_wait_asynccnt)
  __builtin_amdgcn_s_wait_asynccnt((short)N);
#else
  asm volatile("s_wait_asynccnt %0" :: "n"(N) : "memory");
#endif
}

// ---------------- one-time prep kernels ----------------

__global__ void prep_w_kernel(const float* __restrict__ Wih, const float* __restrict__ Whh,
                              _Float16* __restrict__ Bfrag) {
  int idx = blockIdx.x * blockDim.x + threadIdx.x;   // 4096*2048 threads
  int n = idx >> 11;
  int k = idx & 2047;
  float v = (k < Hd) ? Wih[n * Hd + k] : Whh[n * Hd + (k - Hd)];
  Bfrag[frag_idx(n, k, KC_A)] = (_Float16)v;
}

__global__ void prep_bias_kernel(const float* __restrict__ bih, const float* __restrict__ bhh,
                                 float* __restrict__ bias) {
  int i = blockIdx.x * blockDim.x + threadIdx.x;
  if (i < 4 * Hd) bias[i] = bih[i] + bhh[i];
}

__global__ void prep_outw_kernel(const float* __restrict__ outw, _Float16* __restrict__ owfrag) {
  int idx = blockIdx.x * blockDim.x + threadIdx.x;   // 128*1024 threads
  int n = idx >> 10;
  int k = idx & 1023;
  owfrag[frag_idx(n, k, KC_H)] = (_Float16)outw[n * Hd + k];
}

__global__ void x0_kernel(const float* __restrict__ start, const float* __restrict__ embw,
                          const float* __restrict__ embb, float* __restrict__ x0) {
  int j = blockIdx.x * blockDim.x + threadIdx.x;
  if (j >= Hd) return;
  float s = embb[j];
  for (int k = 0; k < Od; ++k) s += start[k] * embw[j * Od + k];
  x0[j] = s;
}

__global__ void init_state_kernel(const float* __restrict__ ench,
                                  const float* __restrict__ w1, const float* __restrict__ b1,
                                  const float* __restrict__ w2, const float* __restrict__ b2,
                                  float* __restrict__ hbuf, float* __restrict__ cbuf) {
  int idx = blockIdx.x * blockDim.x + threadIdx.x;   // 256*1024 threads
  int b = idx >> 10, j = idx & 1023;
  float s1 = b1[j], s2 = b2[j];
  for (int k = 0; k < LATd; ++k) {
    float e = ench[b * LATd + k];
    s1 += e * w1[j * LATd + k];
    s2 += e * w2[j * LATd + k];
  }
  hbuf[idx] = lrelu_(s1);
  cbuf[idx] = lrelu_(s2);
}

__global__ void build_a0_kernel(const float* __restrict__ x0, const float* __restrict__ hbuf,
                                _Float16* __restrict__ Afrag) {
  int idx = blockIdx.x * blockDim.x + threadIdx.x;   // 256*1024 threads
  int b = idx >> 10, j = idx & 1023;
  float xv = x0[j];
  Afrag[frag_idx(b, j, KC_A)]      = (_Float16)(xv > 0.f ? xv : 0.f);   // relu(x) half
  Afrag[frag_idx(b, Hd + j, KC_A)] = (_Float16)hbuf[b * Hd + j];        // h half
}

// ---------------- LSTM step: fused gates GEMM + cell ----------------
// grid = 128 blocks: rm = blk&3 (64 rows), cn = blk>>2 (32 j-cols). 8 waves.
// Wave w: mi = w>>1 (m-tile within block), ji = w&1 (j half). Wave computes all
// FOUR gates for its 16x16 (m,j) tile -> the LSTM cell is wave-local (no LDS
// gate exchange). Operands staged global->LDS with async copies, double-buffered.
// 4 accumulators/wave keeps register pressure low enough that the compiler pins
// them in-place for v_wmma (no accumulator copies / hazard nops — verified).
__global__ void __launch_bounds__(256)
lstm_step_kernel(const _Float16* __restrict__ Acur, _Float16* __restrict__ Anext,
                 const _Float16* __restrict__ Bfrag, const float* __restrict__ bias,
                 float* __restrict__ cbuf, float* __restrict__ hbuf,
                 _Float16* __restrict__ hsfrag_t) {
  __shared__ _Float16 ldsA[2][4][512];   // 8 KB : 4 m-tiles per k-chunk
  __shared__ _Float16 ldsB[2][8][512];   // 16 KB: 8 n-tiles (4 gates x 2 j) per k-chunk
  const int tid = threadIdx.x;
  const int w = tid >> 5, lane = tid & 31;
  const int rm = blockIdx.x & 3;
  const int cn = blockIdx.x >> 2;
  const int mi = w >> 1, ji = w & 1;

  // staging roles (3 async b128 per wave per k-chunk):
  //   B: wave w copies n-tile tt=w  (gate tt>>1, j-half tt&1)
  //   A: wave w copies m-tile (w>>1), 16B-half (w&1)
  const int niW  = (w >> 1) * 64 + cn * 2 + (w & 1);
  const int amt  = w >> 1, hf = w & 1;
  const int miGW = rm * 4 + amt;

  auto stage = [&](int buf, int kc) {
    const _Float16* gb = Bfrag + (size_t)((niW * KC_A + kc) * 32 + lane) * 16;
    _Float16* lb = &ldsB[buf][w][lane * 16];
    async_copy16(gb, lb);
    async_copy16(gb + 8, lb + 8);
    const _Float16* ga = Acur + (size_t)((miGW * KC_A + kc) * 32 + lane) * 16 + hf * 8;
    async_copy16(ga, &ldsA[buf][amt][lane * 16 + hf * 8]);
  };

  v8f acc0 = {}, acc1 = {}, acc2 = {}, acc3 = {};

  stage(0, 0);
  for (int kc = 0; kc < KC_A; ++kc) {
    const int cur = kc & 1;
    if (kc + 1 < KC_A) { stage(cur ^ 1, kc + 1); wait_async<3>(); }
    else               { wait_async<0>(); }
    __syncthreads();

    v16h a  = *(const v16h*)&ldsA[cur][mi][lane * 16];
    v16h b0 = *(const v16h*)&ldsB[cur][0 + ji][lane * 16];
    v16h b1 = *(const v16h*)&ldsB[cur][2 + ji][lane * 16];
    v16h b2 = *(const v16h*)&ldsB[cur][4 + ji][lane * 16];
    v16h b3 = *(const v16h*)&ldsB[cur][6 + ji][lane * 16];
    acc0 = __builtin_amdgcn_wmma_f32_16x16x32_f16(false, a, false, b0, (short)0, acc0, false, false);
    acc1 = __builtin_amdgcn_wmma_f32_16x16x32_f16(false, a, false, b1, (short)0, acc1, false, false);
    acc2 = __builtin_amdgcn_wmma_f32_16x16x32_f16(false, a, false, b2, (short)0, acc2, false, false);
    acc3 = __builtin_amdgcn_wmma_f32_16x16x32_f16(false, a, false, b3, (short)0, acc3, false, false);

    __syncthreads();   // protect buffer reuse before next stage() overwrite
  }

  // wave-local LSTM cell: acc0..acc3 = i,f,g,o pre-activations for this tile
  const int jcol = cn * 32 + ji * 16 + (lane & 15);
  const float bi = bias[jcol];
  const float bf = bias[Hd + jcol];
  const float bg = bias[2 * Hd + jcol];
  const float bo = bias[3 * Hd + jcol];
#pragma unroll
  for (int v = 0; v < 8; ++v) {
    const int brow = rm * 64 + mi * 16 + v + ((lane >> 4) << 3);
    float iv = sigmoidf_(acc0[v] + bi);
    float fv = sigmoidf_(acc1[v] + bf);
    float gv = tanhf(acc2[v] + bg);
    float ov = sigmoidf_(acc3[v] + bo);
    float cp = cbuf[brow * Hd + jcol];
    float cnew = fv * cp + iv * gv;
    float hn = ov * tanhf(cnew);
    cbuf[brow * Hd + jcol] = cnew;
    hbuf[brow * Hd + jcol] = hn;
    Anext[frag_idx(brow, jcol, KC_A)]      = (_Float16)(hn > 0.f ? hn : 0.f);  // relu(x_next)
    Anext[frag_idx(brow, Hd + jcol, KC_A)] = (_Float16)hn;                     // h_next
    hsfrag_t[frag_idx(brow, jcol, KC_H)]   = (_Float16)hn;                     // for out-proj
  }
}

// ---------------- output projection: dec[b,t,:] = hs[t] @ out_w^T + b ----------------
__global__ void __launch_bounds__(256)
outproj_kernel(const _Float16* __restrict__ hsfrag, const _Float16* __restrict__ wfrag,
               const float* __restrict__ outb, float* __restrict__ dec) {
  const int t = blockIdx.x;
  const int tid = threadIdx.x, w = tid >> 5, lane = tid & 31;
  const _Float16* A = hsfrag + (size_t)t * Bsz * Hd;
  const int m0 = w * 2, m1 = w * 2 + 1;
  const _Float16* a0p = A + (size_t)((m0 * KC_H) * 32 + lane) * 16;
  const _Float16* a1p = A + (size_t)((m1 * KC_H) * 32 + lane) * 16;
  for (int nt = 0; nt < 8; ++nt) {
    const _Float16* bp = wfrag + (size_t)((nt * KC_H) * 32 + lane) * 16;
    v8f acc0 = {}, acc1 = {};
    for (int kc = 0; kc < KC_H; ++kc) {
      v16h a0 = *(const v16h*)(a0p + (size_t)kc * 512);
      v16h a1 = *(const v16h*)(a1p + (size_t)kc * 512);
      v16h bb = *(const v16h*)(bp  + (size_t)kc * 512);
      acc0 = __builtin_amdgcn_wmma_f32_16x16x32_f16(false, a0, false, bb, (short)0, acc0, false, false);
      acc1 = __builtin_amdgcn_wmma_f32_16x16x32_f16(false, a1, false, bb, (short)0, acc1, false, false);
    }
    const int o = nt * 16 + (lane & 15);
    const float bo = outb[o];
#pragma unroll
    for (int v = 0; v < 8; ++v) {
      int r0 = m0 * 16 + v + ((lane >> 4) << 3);
      int r1 = m1 * 16 + v + ((lane >> 4) << 3);
      dec[(size_t)r0 * (Tt * Od) + t * Od + o] = acc0[v] + bo;
      dec[(size_t)r1 * (Tt * Od) + t * Od + o] = acc1[v] + bo;
    }
  }
}

// ---------------- small heads (naive f32, negligible FLOPs) ----------------
__global__ void lin_lrelu_kernel(const float* __restrict__ A, const float* __restrict__ W,
                                 const float* __restrict__ bias, float* __restrict__ C,
                                 int K, int J) {
  int idx = blockIdx.x * blockDim.x + threadIdx.x;
  int b = idx / J, j = idx % J;
  if (b >= Bsz) return;
  float s = bias[j];
  for (int k = 0; k < K; ++k) s += A[b * K + k] * W[j * K + k];
  C[idx] = lrelu_(s);
}

__global__ void num_final_kernel(const float* __restrict__ t1, const float* __restrict__ w2,
                                 const float* __restrict__ b2, float* __restrict__ onum) {
  int b = blockIdx.x * blockDim.x + threadIdx.x;
  if (b >= Bsz) return;
  float s = b2[0];
  for (int k = 0; k < Hd; ++k) s += t1[b * Hd + k] * w2[k];
  onum[b] = s > 0.f ? s : 0.f;
}

__global__ void mass_final_kernel(const float* __restrict__ m2, const float* __restrict__ w3,
                                  const float* __restrict__ b3, float* __restrict__ omass) {
  int b = blockIdx.x * blockDim.x + threadIdx.x;
  if (b >= Bsz) return;
  float s0 = b3[0], s1 = b3[1];
  for (int k = 0; k < Hd; ++k) {
    float m = m2[b * Hd + k];
    s0 += m * w3[k];
    s1 += m * w3[Hd + k];
  }
  float mx = fmaxf(s0, s1);
  float e0 = expf(s0 - mx), e1 = expf(s1 - mx);
  float d = e0 + e1;
  omass[b * 2]     = e0 / d;
  omass[b * 2 + 1] = e1 / d;
}

__global__ void finalize_kernel(const float* __restrict__ hbuf, const float* __restrict__ cbuf,
                                float* __restrict__ oh, float* __restrict__ oc) {
  int i = blockIdx.x * blockDim.x + threadIdx.x;
  if (i < Bsz * Hd) { oh[i] = hbuf[i]; oc[i] = cbuf[i]; }
}

// ---------------- launch ----------------
extern "C" void kernel_launch(void* const* d_in, const int* in_sizes, int n_in,
                              void* d_out, int out_size, void* d_ws, size_t ws_size,
                              hipStream_t stream) {
  (void)in_sizes; (void)n_in; (void)out_size; (void)ws_size;
  const float* enc_out   = (const float*)d_in[0];
  const float* enc_hid   = (const float*)d_in[1];
  const float* start_tok = (const float*)d_in[2];
  const float* l2h_w  = (const float*)d_in[3];
  const float* l2h_b  = (const float*)d_in[4];
  const float* l2h2_w = (const float*)d_in[5];
  const float* l2h2_b = (const float*)d_in[6];
  const float* emb_w  = (const float*)d_in[7];
  const float* emb_b  = (const float*)d_in[8];
  const float* W_ih   = (const float*)d_in[9];
  const float* W_hh   = (const float*)d_in[10];
  const float* b_ih   = (const float*)d_in[11];
  const float* b_hh   = (const float*)d_in[12];
  const float* out_w  = (const float*)d_in[13];
  const float* out_b  = (const float*)d_in[14];
  const float* fs_w   = (const float*)d_in[15];
  const float* fs_b   = (const float*)d_in[16];
  const float* fs2_w  = (const float*)d_in[17];
  const float* fs2_b  = (const float*)d_in[18];
  const float* fm_w   = (const float*)d_in[19];
  const float* fm_b   = (const float*)d_in[20];
  const float* fm2_w  = (const float*)d_in[21];
  const float* fm2_b  = (const float*)d_in[22];
  const float* fm3_w  = (const float*)d_in[23];
  const float* fm3_b  = (const float*)d_in[24];

  // workspace carve-out
  char* base = (char*)d_ws;
  size_t off = 0;
  auto alloc = [&](size_t bytes) { char* p = base + off; off += (bytes + 255) & ~(size_t)255; return p; };
  float*    bias   = (float*)alloc((size_t)4 * Hd * 4);
  float*    cbuf   = (float*)alloc((size_t)Bsz * Hd * 4);
  float*    hbuf   = (float*)alloc((size_t)Bsz * Hd * 4);
  float*    x0     = (float*)alloc((size_t)Hd * 4);
  float*    t1     = (float*)alloc((size_t)Bsz * Hd * 4);
  float*    m2a    = (float*)alloc((size_t)Bsz * Hd * 4);
  float*    m2b    = (float*)alloc((size_t)Bsz * Hd * 4);
  _Float16* Afrag0 = (_Float16*)alloc((size_t)Bsz * 2 * Hd * 2);
  _Float16* Afrag1 = (_Float16*)alloc((size_t)Bsz * 2 * Hd * 2);
  _Float16* Bfrag  = (_Float16*)alloc((size_t)2 * Hd * 4 * Hd * 2);   // 16 MB
  _Float16* owfrag = (_Float16*)alloc((size_t)Hd * Od * 2);
  _Float16* hsfrag = (_Float16*)alloc((size_t)Tt * Bsz * Hd * 2);     // 50 MB

  // output regions (concatenated, reference return order)
  float* outp  = (float*)d_out;
  float* dec   = outp;                                  // [256,100,128]
  float* hT    = outp + (size_t)Bsz * Tt * Od;          // [1,256,1024]
  float* cT    = hT + (size_t)Bsz * Hd;                 // [1,256,1024]
  float* onum  = cT + (size_t)Bsz * Hd;                 // [256,1]
  float* omass = onum + Bsz;                            // [256,2]

  // one-time prep
  prep_w_kernel<<<(4 * Hd * 2 * Hd) / 256, 256, 0, stream>>>(W_ih, W_hh, Bfrag);
  prep_bias_kernel<<<16, 256, 0, stream>>>(b_ih, b_hh, bias);
  prep_outw_kernel<<<(Od * Hd) / 256, 256, 0, stream>>>(out_w, owfrag);
  x0_kernel<<<4, 256, 0, stream>>>(start_tok, emb_w, emb_b, x0);
  init_state_kernel<<<(Bsz * Hd) / 256, 256, 0, stream>>>(enc_hid, l2h_w, l2h_b, l2h2_w, l2h2_b, hbuf, cbuf);
  build_a0_kernel<<<(Bsz * Hd) / 256, 256, 0, stream>>>(x0, hbuf, Afrag0);

  // recurrence: 100 sequential WMMA step kernels, ping-pong A fragments
  _Float16* Aping[2] = {Afrag0, Afrag1};
  for (int t = 0; t < Tt; ++t) {
    lstm_step_kernel<<<128, 256, 0, stream>>>(Aping[t & 1], Aping[(t + 1) & 1], Bfrag, bias,
                                              cbuf, hbuf, hsfrag + (size_t)t * Bsz * Hd);
  }

  // output projection
  outproj_kernel<<<Tt, 256, 0, stream>>>(hsfrag, owfrag, out_b, dec);

  // heads
  lin_lrelu_kernel<<<(Bsz * Hd) / 256, 256, 0, stream>>>(enc_out, fs_w, fs_b, t1, LATd, Hd);
  num_final_kernel<<<1, 256, 0, stream>>>(t1, fs2_w, fs2_b, onum);
  lin_lrelu_kernel<<<(Bsz * Hd) / 256, 256, 0, stream>>>(enc_out, fm_w, fm_b, m2a, LATd, Hd);
  lin_lrelu_kernel<<<(Bsz * Hd) / 256, 256, 0, stream>>>(m2a, fm2_w, fm2_b, m2b, Hd, Hd);
  mass_final_kernel<<<1, 256, 0, stream>>>(m2b, fm3_w, fm3_b, omass);

  // hT / cT
  finalize_kernel<<<(Bsz * Hd) / 256, 256, 0, stream>>>(hbuf, cbuf, hT, cT);
}
